// MultiHeadAttention_78142634983539
// MI455X (gfx1250) — matrix operand order repaired and checked by reading, not compiled, not run on hardware
//
#include <hip/hip_runtime.h>
#include <hip/hip_bf16.h>
#include <cstdint>

// ---------------------------------------------------------------------------
// MHA forward for MI455X (gfx1250, wave32).
// All GEMM stages run on v_wmma_f32_16x16x32_bf16 (K=32 bf16, fp32 accum).
// Projection AND attention kernels stream shared operand tiles into LDS via
// the Tensor Data Mover (tensor_load_to_lds + s_wait_tensorcnt), double
// buffered across workgroup barriers. Softmax is fp32 VALU with cross-lane
// reductions; the causal mask is analytic (mask input unused).
// Workspace layout (bytes), total ~72 MB (assumes ws_size >= 75,497,472):
//   [0,   8M)  : wq,wk,wv,wo as bf16 (4 * 2^20 elements)
//   [8M, 24M)  : Qh  bf16 [B,H,S,DH]
//   [24M,40M)  : Kh  bf16 [B,H,S,DH]
//   [40M,56M)  : Vh  bf16 [B,H,S,DH]
//   [56M,72M)  : ctx bf16 [B,S,D]
// ---------------------------------------------------------------------------

#define BB 4
#define SS 2048
#define DD 1024
#define HH 16
#define DH 64

typedef __attribute__((ext_vector_type(16))) __bf16 v16bf;
typedef __attribute__((ext_vector_type(8)))  float  v8f;

#if defined(__has_builtin)
#if __has_builtin(__builtin_amdgcn_tensor_load_to_lds) && \
    __has_builtin(__builtin_amdgcn_s_wait_tensorcnt)
#define HAVE_TDM 1
#endif
#endif
#ifndef HAVE_TDM
#define HAVE_TDM 0
#endif

static __device__ inline v8f wmma_bf16(v16bf a, v16bf b, v8f c) {
  return __builtin_amdgcn_wmma_f32_16x16x32_bf16(
      /*neg_a=*/false, a, /*neg_b=*/false, b,
      /*c_mod=*/(short)0, c, /*reuse_a=*/false, /*reuse_b=*/false);
}

// Generic LDS pointer -> LDS byte offset (addr[31:0] per flat-aperture rules).
static __device__ inline unsigned lds_offset_of(const void* p) {
  return (unsigned)(uintptr_t)p;
}

// ---------------------------------------------------------------------------
// Tensor Data Mover: DMA a 2-D bf16 tile (rows x cols, row stride in elems)
// from global into LDS at lds_off bytes. Per-wave issue; tracked by TENSORcnt.
// (6-arg builtin: D# groups 0..3 + spare + cpol.)
// ---------------------------------------------------------------------------
static __device__ inline void tdm_load_tile_bf16(unsigned lds_off,
                                                 const __bf16* __restrict__ gsrc,
                                                 unsigned rows, unsigned cols,
                                                 unsigned row_stride_elems) {
#if HAVE_TDM
  typedef unsigned int v4u __attribute__((ext_vector_type(4)));
  typedef int v8i __attribute__((ext_vector_type(8)));
  typedef int v4i __attribute__((ext_vector_type(4)));
  unsigned long long ga = (unsigned long long)(uintptr_t)gsrc;

  // D# group 0: count=1, lds_addr, global_addr, type=2.
  v4u g0;
  g0[0] = 1u;
  g0[1] = lds_off;
  g0[2] = (unsigned)(ga & 0xFFFFFFFFu);
  g0[3] = (unsigned)((ga >> 32) & 0x1FFFFFFu) | (2u << 30);

  // D# group 1: data_size=1 (2B); 2-D tile cols x rows; tensor dims/stride.
  v8i g1;
  g1[0] = (int)(1u << 16);
  g1[1] = (int)((row_stride_elems & 0xFFFFu) << 16);
  g1[2] = (int)(((row_stride_elems >> 16) & 0xFFFFu) | ((rows & 0xFFFFu) << 16));
  g1[3] = (int)(((rows >> 16) & 0xFFFFu) | ((cols & 0xFFFFu) << 16));
  g1[4] = (int)(rows & 0xFFFFu);
  g1[5] = (int)row_stride_elems;
  g1[6] = 0;
  g1[7] = 0;

  v4i gz4 = {0, 0, 0, 0};
  v8i gz8 = {0, 0, 0, 0, 0, 0, 0, 0};
  __builtin_amdgcn_tensor_load_to_lds(g0, g1, gz4, gz4, gz8, 0);
#else
  (void)lds_off; (void)gsrc; (void)rows; (void)cols; (void)row_stride_elems;
#endif
}

static __device__ inline void tdm_wait0() {
#if HAVE_TDM
  __builtin_amdgcn_s_wait_tensorcnt(0);
#endif
}

// Fallback staging when the TDM builtin is unavailable.
static __device__ inline void stage_tile_fallback(__bf16* __restrict__ dst,
                                                  const __bf16* __restrict__ src,
                                                  int rows, int cols, int ld,
                                                  int lane) {
  int total = rows * cols / 2;  // uint elements
  for (int i = lane; i < total; i += 32) {
    int r = i / (cols / 2), c = i - r * (cols / 2);
    ((unsigned*)dst)[i] = *(const unsigned*)(src + (size_t)r * ld + 2 * c);
  }
}

// ---- Fragment loaders (contiguous k order: k = 16*(lane/16) + s) ----------------------
static __device__ inline v16bf load_a_bf16(const __bf16* __restrict__ M, int ld,
                                           int row0, int k0, int lane) {
  int m = lane & 15, hi = lane >> 4;
  const __bf16* r = M + (size_t)(row0 + m) * ld + k0;
  v16bf a;
#pragma unroll
  for (int v = 0; v < 8; ++v) {
    int kk = hi * 16 + 2 * v;
    a[2 * v]     = r[kk];
    a[2 * v + 1] = r[kk + 1];
  }
  return a;
}

static __device__ inline v16bf load_a_f32(const float* __restrict__ M, int ld,
                                          int row0, int k0, int lane) {
  int m = lane & 15, hi = lane >> 4;
  const float* r = M + (size_t)(row0 + m) * ld + k0;
  v16bf a;
#pragma unroll
  for (int v = 0; v < 8; ++v) {
    int kk = hi * 16 + 2 * v;
    a[2 * v]     = (__bf16)r[kk];
    a[2 * v + 1] = (__bf16)r[kk + 1];
  }
  return a;
}

// B[k][n] = M[n][k] (M row-major): weight W for x*W^T, K-matrix for Q*K^T.
static __device__ inline v16bf load_bT_rowmaj_bf16(const __bf16* __restrict__ M,
                                                   int ld, int n0, int k0, int lane) {
  int n = lane & 15, hi = lane >> 4;
  const __bf16* r = M + (size_t)(n0 + n) * ld + k0 + hi * 16;
  v16bf b;
#pragma unroll
  for (int v = 0; v < 8; ++v) {
    b[2 * v]     = r[2 * v];
    b[2 * v + 1] = r[2 * v + 1];
  }
  return b;
}

// B[k][n] = M[k][n] (M row-major): V in P*V.
static __device__ inline v16bf load_b_rowmaj_bf16(const __bf16* __restrict__ M,
                                                  int ld, int k0, int n0, int lane) {
  int n = lane & 15, hi = lane >> 4;
  v16bf b;
#pragma unroll
  for (int v = 0; v < 8; ++v) {
    int kk = k0 + hi * 16 + 2 * v;
    const __bf16* p = M + (size_t)kk * ld + n0 + n;
    b[2 * v]     = p[0];
    b[2 * v + 1] = p[ld];
  }
  return b;
}

// ---------------------------------------------------------------------------
// Kernel 0: fp32 -> bf16 weight conversion (4 weights of D*D = 2^20 each)
// ---------------------------------------------------------------------------
__global__ __launch_bounds__(256) void cvt_weights_kernel(
    const float* __restrict__ wq, const float* __restrict__ wk,
    const float* __restrict__ wv, const float* __restrict__ wo,
    __bf16* __restrict__ dst) {
  size_t i = (size_t)blockIdx.x * 256 + threadIdx.x;
  int sel = (int)(i >> 20);
  size_t off = i & ((1u << 20) - 1);
  const float* src = (sel == 0) ? wq : (sel == 1) ? wk : (sel == 2) ? wv : wo;
  dst[i] = (__bf16)src[off];
}

// ---------------------------------------------------------------------------
// Shared GEMM body: 4 waves/WG; each wave owns a 16-row A tile; all waves share
// a TDM-staged 64x32 weight tile per K-step (double buffered in LDS).
// ---------------------------------------------------------------------------
struct GemmTileArgs {
  int lane, wave, row0, nb0;
};

template <typename LoadA>
static __device__ inline void gemm_wtile_body(__bf16 (&wtile)[2][64 * 32],
                                              const __bf16* __restrict__ w,
                                              GemmTileArgs ar, v8f (&acc)[4],
                                              LoadA loadA) {
  const __bf16* wsrc = w + (size_t)(ar.nb0 + ar.wave * 16) * DD;
  unsigned lds0 = lds_offset_of(&wtile[0][ar.wave * 16 * 32]);
  unsigned lds1 = lds_offset_of(&wtile[1][ar.wave * 16 * 32]);

#if HAVE_TDM
  tdm_load_tile_bf16(lds0, wsrc, 16, 32, DD);
#else
  stage_tile_fallback(&wtile[0][ar.wave * 16 * 32], wsrc, 16, 32, DD, ar.lane);
#endif

  const int NSTEP = DD / 32;
  for (int s = 0; s < NSTEP; ++s) {
    int cur = s & 1;
    tdm_wait0();            // this wave's slice of buf[cur] has landed
    __syncthreads();        // all slices landed; buf[cur^1] free to overwrite

    if (s + 1 < NSTEP) {
      const __bf16* nxt = wsrc + (s + 1) * 32;
#if HAVE_TDM
      tdm_load_tile_bf16(cur ? lds0 : lds1, nxt, 16, 32, DD);
#else
      stage_tile_fallback(&wtile[cur ^ 1][ar.wave * 16 * 32], nxt, 16, 32, DD,
                          ar.lane);
#endif
    }

    v16bf a = loadA(s * 32);
    const __bf16* wt = &wtile[cur][0];
#pragma unroll
    for (int t = 0; t < 4; ++t) {
      v16bf b = load_bT_rowmaj_bf16(wt, 32, t * 16, 0, ar.lane);
      acc[t] = wmma_bf16(a, b, acc[t]);
    }
  }
}

// ---------------------------------------------------------------------------
// Kernel 1: fused QKV projection.  y = x @ W^T + b, reshaped to [B,H,S,DH].
// grid = (BS/64, D/64, 3), block = 128 (4 waves, one 16x64 tile each).
// ---------------------------------------------------------------------------
__global__ __launch_bounds__(128) void proj_qkv_kernel(
    const float* __restrict__ xq, const float* __restrict__ xk,
    const float* __restrict__ xv,
    const __bf16* __restrict__ wqb, const __bf16* __restrict__ wkb,
    const __bf16* __restrict__ wvb,
    const float* __restrict__ bq, const float* __restrict__ bk,
    const float* __restrict__ bv,
    __bf16* __restrict__ qh, __bf16* __restrict__ kh, __bf16* __restrict__ vh) {
  __shared__ __bf16 wtile[2][64 * 32];

  GemmTileArgs ar;
  ar.lane = threadIdx.x & 31;
  ar.wave = threadIdx.x >> 5;
  ar.row0 = (blockIdx.x * 4 + ar.wave) * 16;  // over B*S
  ar.nb0  = blockIdx.y * 64;                  // over D
  int z = blockIdx.z;

  const float*  x    = (z == 0) ? xq  : (z == 1) ? xk  : xv;
  const __bf16* w    = (z == 0) ? wqb : (z == 1) ? wkb : wvb;
  const float*  bias = (z == 0) ? bq  : (z == 1) ? bk  : bv;
  __bf16*       outh = (z == 0) ? qh  : (z == 1) ? kh  : vh;

  int n = ar.lane & 15, hi = ar.lane >> 4;

  v8f acc[4];
#pragma unroll
  for (int t = 0; t < 4; ++t) {
    float bn = bias[ar.nb0 + t * 16 + n];
#pragma unroll
    for (int v = 0; v < 8; ++v) acc[t][v] = bn;
  }

  gemm_wtile_body(wtile, w, ar, acc, [&](int k0) {
    return load_a_f32(x, DD, ar.row0, k0, ar.lane);
  });

#pragma unroll
  for (int t = 0; t < 4; ++t) {
    int ncol = ar.nb0 + t * 16 + n;
    int h = ncol >> 6, dh = ncol & 63;
#pragma unroll
    for (int v = 0; v < 8; ++v) {
      int r = ar.row0 + v + 8 * hi;
      int b = r >> 11, s = r & 2047;   // S = 2048
      outh[(((size_t)b * HH + h) * SS + s) * DH + dh] = (__bf16)acc[t][v];
    }
  }
}

// ---------------------------------------------------------------------------
// Kernel 2: causal flash-attention. 4 waves/WG handle 4 consecutive 16-query
// tiles of one (b,h); the 32-key K and V blocks are TDM-staged into LDS
// (waves 0/1 DMA K halves, waves 2/3 DMA V halves), double buffered.
// grid = (S/64, B*H), block = 128.
// ---------------------------------------------------------------------------
__global__ __launch_bounds__(128) void attn_kernel(
    const __bf16* __restrict__ qh, const __bf16* __restrict__ kh,
    const __bf16* __restrict__ vh, __bf16* __restrict__ ctx) {
  __shared__ __bf16 Kt[2][32 * 64];   // 8 KB: 32 keys x DH
  __shared__ __bf16 Vt[2][32 * 64];   // 8 KB
  __shared__ __bf16 Pt[4][16 * 32];   // 4 KB: per-wave P transpose staging

  int lane = threadIdx.x & 31;
  int wave = threadIdx.x >> 5;
  int n = lane & 15, hi = lane >> 4;
  int it = blockIdx.x * 4 + wave;    // this wave's query tile
  int bh = blockIdx.y;
  int b = bh >> 4, h = bh & 15;      // H = 16

  const __bf16* Q = qh + (size_t)bh * SS * DH;
  const __bf16* K = kh + (size_t)bh * SS * DH;
  const __bf16* V = vh + (size_t)bh * SS * DH;

  v16bf aq0 = load_a_bf16(Q, DH, it * 16, 0, lane);
  v16bf aq1 = load_a_bf16(Q, DH, it * 16, 32, lane);

  // Staging role: wave 0/1 -> K rows [0,16)/[16,32) of block; 2/3 -> V same.
  int half = wave & 1;
  const __bf16* ssrc = ((wave < 2) ? K : V) + (size_t)(half * 16) * DH;
  __bf16* sb0 = (wave < 2) ? &Kt[0][half * 16 * 64] : &Vt[0][half * 16 * 64];
  __bf16* sb1 = (wave < 2) ? &Kt[1][half * 16 * 64] : &Vt[1][half * 16 * 64];
  unsigned slds0 = lds_offset_of(sb0);
  unsigned slds1 = lds_offset_of(sb1);

  v8f o0 = {}, o1 = {}, o2 = {}, o3 = {};
  float mi[8], li[8];
#pragma unroll
  for (int v = 0; v < 8; ++v) { mi[v] = -3.0e38f; li[v] = 0.0f; }

  const float scale = 0.125f;  // 1/sqrt(64)
  // Uniform trip count for the whole WG (causal bound of its last q-tile);
  // fully-masked blocks for earlier tiles are numerically inert.
  int nb = ((blockIdx.x * 4 + 3) * 16 + 15) / 32 + 1;

#if HAVE_TDM
  tdm_load_tile_bf16(slds0, ssrc, 16, 64, DH);
#else
  stage_tile_fallback(sb0, ssrc, 16, 64, DH, lane);
#endif

  for (int jb = 0; jb < nb; ++jb) {
    int cur = jb & 1;
    tdm_wait0();            // this wave's DMA slice of buf[cur] landed
    __syncthreads();        // whole K/V block staged; buf[cur^1] reusable

    if (jb + 1 < nb) {
      const __bf16* nxt = ssrc + (size_t)(jb + 1) * 32 * DH;
#if HAVE_TDM
      tdm_load_tile_bf16(cur ? slds0 : slds1, nxt, 16, 64, DH);
#else
      stage_tile_fallback(cur ? sb0 : sb1, nxt, 16, 64, DH, lane);
#endif
    }

    int k0 = jb * 32;
    const __bf16* Kc = &Kt[cur][0];
    const __bf16* Vc = &Vt[cur][0];

    // ---- S = Q K^T over 32 keys (two 16x16 tiles, K-dim = DH = 64) ----
    v8f z = {};
    v8f s0 = wmma_bf16(aq0, load_bT_rowmaj_bf16(Kc, 64, 0, 0, lane), z);
    s0     = wmma_bf16(aq1, load_bT_rowmaj_bf16(Kc, 64, 0, 32, lane), s0);
    v8f s1 = wmma_bf16(aq0, load_bT_rowmaj_bf16(Kc, 64, 16, 0, lane), z);
    s1     = wmma_bf16(aq1, load_bT_rowmaj_bf16(Kc, 64, 16, 32, lane), s1);

    // ---- online softmax (fp32, per-row across 16-lane half groups) ----
#pragma unroll
    for (int v = 0; v < 8; ++v) {
      int m = v + 8 * hi;
      int qidx = it * 16 + m;
      float e0 = s0[v] * scale;
      float e1 = s1[v] * scale;
      if (k0 + n > qidx)      e0 = -3.0e38f;  // causal mask
      if (k0 + 16 + n > qidx) e1 = -3.0e38f;

      float rm = fmaxf(e0, e1);
#pragma unroll
      for (int off = 1; off < 16; off <<= 1)
        rm = fmaxf(rm, __shfl_xor(rm, off, 32));

      float mn = fmaxf(mi[v], rm);
      float corr = __expf(mi[v] - mn);
      mi[v] = mn;

      float p0 = __expf(e0 - mn);
      float p1 = __expf(e1 - mn);
      float rs = p0 + p1;
#pragma unroll
      for (int off = 1; off < 16; off <<= 1)
        rs += __shfl_xor(rs, off, 32);

      li[v] = li[v] * corr + rs;
      o0[v] *= corr; o1[v] *= corr; o2[v] *= corr; o3[v] *= corr;

      Pt[wave][m * 32 + n]      = (__bf16)p0;
      Pt[wave][m * 32 + 16 + n] = (__bf16)p1;
    }
    __syncthreads();  // uniform; orders per-wave P store -> load

    // ---- O += P V (16x32 P, 32x64 V -> four 16x16 tiles) ----
    v16bf ap = load_a_bf16(&Pt[wave][0], 32, 0, 0, lane);
    o0 = wmma_bf16(ap, load_b_rowmaj_bf16(Vc, 64, 0, 0,  lane), o0);
    o1 = wmma_bf16(ap, load_b_rowmaj_bf16(Vc, 64, 0, 16, lane), o1);
    o2 = wmma_bf16(ap, load_b_rowmaj_bf16(Vc, 64, 0, 32, lane), o2);
    o3 = wmma_bf16(ap, load_b_rowmaj_bf16(Vc, 64, 0, 48, lane), o3);
  }

  // ---- normalize and store ctx as bf16 [B,S,D] ----
#pragma unroll
  for (int v = 0; v < 8; ++v) {
    float inv = (li[v] > 0.0f) ? (1.0f / li[v]) : 0.0f;
    int m = v + 8 * hi;
    size_t base = ((size_t)b * SS + it * 16 + m) * DD + h * 64 + n;
    ctx[base +  0] = (__bf16)(o0[v] * inv);
    ctx[base + 16] = (__bf16)(o1[v] * inv);
    ctx[base + 32] = (__bf16)(o2[v] * inv);
    ctx[base + 48] = (__bf16)(o3[v] * inv);
  }
}

// ---------------------------------------------------------------------------
// Kernel 3: output projection  out = ctx @ Wo^T + bo  (fp32 out)
// grid = (BS/64, D/64), block = 128 (4 waves)
// ---------------------------------------------------------------------------
__global__ __launch_bounds__(128) void oproj_kernel(
    const __bf16* __restrict__ ctx, const __bf16* __restrict__ wob,
    const float* __restrict__ bo, float* __restrict__ out) {
  __shared__ __bf16 wtile[2][64 * 32];

  GemmTileArgs ar;
  ar.lane = threadIdx.x & 31;
  ar.wave = threadIdx.x >> 5;
  ar.row0 = (blockIdx.x * 4 + ar.wave) * 16;
  ar.nb0  = blockIdx.y * 64;
  int n = ar.lane & 15, hi = ar.lane >> 4;

  v8f acc[4];
#pragma unroll
  for (int t = 0; t < 4; ++t) {
    float bn = bo[ar.nb0 + t * 16 + n];
#pragma unroll
    for (int v = 0; v < 8; ++v) acc[t][v] = bn;
  }

  gemm_wtile_body(wtile, wob, ar, acc, [&](int k0) {
    return load_a_bf16(ctx, DD, ar.row0, k0, ar.lane);
  });

#pragma unroll
  for (int t = 0; t < 4; ++t)
#pragma unroll
    for (int v = 0; v < 8; ++v) {
      size_t r = (size_t)(ar.row0 + v + 8 * hi);
      out[r * DD + ar.nb0 + t * 16 + n] = acc[t][v];
    }
}

// ---------------------------------------------------------------------------
extern "C" void kernel_launch(void* const* d_in, const int* in_sizes, int n_in,
                              void* d_out, int out_size, void* d_ws, size_t ws_size,
                              hipStream_t stream) {
  (void)in_sizes; (void)n_in; (void)out_size; (void)ws_size;
  const float* q    = (const float*)d_in[0];
  const float* k    = (const float*)d_in[1];
  const float* v    = (const float*)d_in[2];
  // d_in[3] = mask (causal) — applied analytically, not read
  const float* wq   = (const float*)d_in[4];
  const float* bq   = (const float*)d_in[5];
  const float* wk   = (const float*)d_in[6];
  const float* bk   = (const float*)d_in[7];
  const float* wv   = (const float*)d_in[8];
  const float* bv   = (const float*)d_in[9];
  const float* wo   = (const float*)d_in[10];
  const float* bo   = (const float*)d_in[11];
  float* out = (float*)d_out;

  const size_t WSZ = (size_t)DD * DD;        // 2^20
  const size_t ACT = (size_t)BB * SS * DD;   // 8M elements

  __bf16* wbf = (__bf16*)d_ws;
  __bf16* wqb = wbf;
  __bf16* wkb = wbf + WSZ;
  __bf16* wvb = wbf + 2 * WSZ;
  __bf16* wob = wbf + 3 * WSZ;
  __bf16* qh  = wbf + 4 * WSZ;
  __bf16* kh  = qh + ACT;
  __bf16* vh  = kh + ACT;
  __bf16* ctx = vh + ACT;

  cvt_weights_kernel<<<(4 * WSZ) / 256, 256, 0, stream>>>(wq, wk, wv, wo, wbf);

  dim3 gproj((BB * SS) / 64, DD / 64, 3);
  proj_qkv_kernel<<<gproj, 128, 0, stream>>>(q, k, v, wqb, wkb, wvb,
                                             bq, bk, bv, qh, kh, vh);

  dim3 gattn(SS / 64, BB * HH);
  attn_kernel<<<gattn, 128, 0, stream>>>(qh, kh, vh, ctx);

  dim3 goproj((BB * SS) / 64, DD / 64);
  oproj_kernel<<<goproj, 128, 0, stream>>>(ctx, wob, bo, out);
}
